// Attention_33371895889954
// MI455X (gfx1250) — matrix operand order repaired and checked by reading, not compiled, not run on hardware
//
#include <hip/hip_runtime.h>
#include <hip/hip_bf16.h>
#include <stdint.h>

#define DIM    512
#define DH     32
#define HEADS  16
#define WSZ    8
#define PPW    64      // pixels per window
#define NWIN   512     // 8 batches * 8 * 8 windows
#define IMG    64
#define EPS    1e-5f
#define SCALE_Q 0.1767766952966369f   // 32^-0.5

// padded LDS row strides (halves): multiples of 8 (16B align) and chosen so
// 16-lane column reads spread over all 64 LDS banks (stride/2 mod 64 != 0).
#define XK_LD  520    // 512-channel window rows
#define KK_LD  40     // K rows (DH=32)
#define VK_LD  72     // V rows (PPW=64)
#define AK_LD  72     // attn-prob rows (PPW=64)

typedef __attribute__((ext_vector_type(16))) _Float16 v16h;
typedef __attribute__((ext_vector_type(8)))  float    v8f;

union H16 {
  v16h v;
  _Float16 h[16];
  uint4 q[2];
};

__device__ __forceinline__ v8f wmma_f16(v16h a, v16h b, v8f c) {
  // v_wmma_f32_16x16x32_f16: D = A(16x32) * B(32x16) + C(16x16 f32)
  return __builtin_amdgcn_wmma_f32_16x16x32_f16(
      /*neg_a=*/false, a, /*neg_b=*/false, b,
      /*c_mod=*/(short)0, c, /*reuse_a=*/false, /*reuse_b=*/false);
}

// Async global->LDS b128 copy (per-lane), tracked by ASYNCcnt.
// LDS byte offset = low 32 bits of the flat shared-aperture address.
__device__ __forceinline__ void async_copy_b128(void* lds_dst, const void* gsrc) {
  uint32_t ldsoff = (uint32_t)(uintptr_t)lds_dst;
  asm volatile("global_load_async_to_lds_b128 %0, %1, off"
               :: "v"(ldsoff), "v"(gsrc)
               : "memory");
}
__device__ __forceinline__ void wait_async() {
  asm volatile("s_wait_asynccnt 0x0" ::: "memory");
}

// A-matrix 16x32 f16 lane layout (ISA 7.12.2):
//   lanes 0-15 : row M=lane,    K in [0,8) then [16,24)
//   lanes 16-31: row M=lane-16, K in [8,16) then [24,32)
// Source row-major [M][K], leading dim ldk (halves). Two b128 loads.
__device__ __forceinline__ v16h load_a16x32(const _Float16* base, int ldk,
                                            int lane, int row0, int k0) {
  int m  = row0 + (lane & 15);
  int kb = k0 + ((lane >> 4) << 3);
  const _Float16* p = base + (size_t)m * ldk + kb;
  H16 r;
  r.q[0] = *(const uint4*)(p);
  r.q[1] = *(const uint4*)(p + 16);
  return r.v;
}

// B-matrix 32x16 f16 lane layout:
//   lane holds column N=(lane&15), K in [8h,8h+8) then [16+8h,24+8h), h=lane>>4.
// Source stored column-major: [N][K], leading dim ldk. Two b128 loads.
__device__ __forceinline__ v16h load_b32x16(const _Float16* colmaj, int ldk,
                                            int lane, int n0, int k0) {
  int n  = n0 + (lane & 15);
  int kb = k0 + ((lane >> 4) << 3);
  const _Float16* p = colmaj + (size_t)n * ldk + kb;
  H16 r;
  r.q[0] = *(const uint4*)(p);
  r.q[1] = *(const uint4*)(p + 16);
  return r.v;
}

// ---------------------------------------------------------------- DPB MLP ---
__global__ void dpb_kernel(const float* __restrict__ w1, const float* __restrict__ b1,
                           const float* __restrict__ g1, const float* __restrict__ be1,
                           const float* __restrict__ w2, const float* __restrict__ b2,
                           const float* __restrict__ g2, const float* __restrict__ be2,
                           const float* __restrict__ w3, const float* __restrict__ b3,
                           float* __restrict__ rel_bias) {
  __shared__ float bias[289];
  int t = threadIdx.x;
  if (t < 289) {
    float gy = (float)(t / 17 - WSZ);
    float gx = (float)(t % 17 - WSZ);
    float h1[128];
    float s = 0.f, ss = 0.f;
    for (int k = 0; k < 128; ++k) {
      float v = w1[2 * k] * gy + w1[2 * k + 1] * gx + b1[k];
      h1[k] = v; s += v; ss += v * v;
    }
    float m = s * (1.f / 128.f);
    float inv = rsqrtf(ss * (1.f / 128.f) - m * m + EPS);
    for (int k = 0; k < 128; ++k)
      h1[k] = fmaxf((h1[k] - m) * inv * g1[k] + be1[k], 0.f);
    float h2[128];
    s = 0.f; ss = 0.f;
    for (int k = 0; k < 128; ++k) {
      float v = b2[k];
      const float* wr = w2 + 128 * k;
      for (int j = 0; j < 128; ++j) v += wr[j] * h1[j];
      h2[k] = v; s += v; ss += v * v;
    }
    m = s * (1.f / 128.f);
    inv = rsqrtf(ss * (1.f / 128.f) - m * m + EPS);
    float o = b3[0];
    for (int k = 0; k < 128; ++k)
      o += w3[k] * fmaxf((h2[k] - m) * inv * g2[k] + be2[k], 0.f);
    bias[t] = o;
  }
  __syncthreads();
  for (int idx = t; idx < PPW * PPW; idx += blockDim.x) {
    int i = idx >> 6, j = idx & 63;
    int yi = i >> 3, xi = i & 7, yj = j >> 3, xj = j & 7;
    rel_bias[idx] = bias[(yi - yj + WSZ - 1) * (2 * WSZ - 1) + (xi - xj + WSZ - 1)];
  }
}

// ------------------------------------------------------------ f32 -> f16 ---
__global__ void cvt_f16(const float* __restrict__ in, _Float16* __restrict__ out, int n) {
  int i = blockIdx.x * blockDim.x + threadIdx.x;
  if (i < n) out[i] = (_Float16)in[i];
}

// ---------------------------------------------- channel LN + window split ---
// x: [b, 512, 64, 64]; output xn: [win][pixel][ch] f16 (ch contiguous).
__global__ void ln_kernel(const float* __restrict__ x, const float* __restrict__ g,
                          const float* __restrict__ bch, _Float16* __restrict__ xn) {
  int pid = blockIdx.x * blockDim.x + threadIdx.x;  // 0..32767 pixels
  int bb = pid >> 12;
  int hw = pid & 4095;
  int hy = hw >> 6, wx = hw & 63;
  const float* px = x + (size_t)bb * DIM * 4096 + hy * 64 + wx;
  float s = 0.f, ss = 0.f;
  for (int c = 0; c < DIM; ++c) {
    float v = px[(size_t)c * 4096];
    s += v; ss += v * v;
  }
  float m = s * (1.f / DIM);
  float var = ss * (1.f / DIM) - m * m;
  float inv = 1.0f / (var + EPS);  // faithful: reference divides by (var+eps), no sqrt
  int n = (bb << 6) | ((hy >> 3) << 3) | (wx >> 3);
  int p = ((hy & 7) << 3) | (wx & 7);
  _Float16* dst = xn + ((size_t)n * PPW + p) * DIM;
  for (int c = 0; c < DIM; ++c)
    dst[c] = (_Float16)((px[(size_t)c * 4096] - m) * inv * g[c] + bch[c]);
}

// ---------------------------------------------------------- QKV GEMM -------
// One block per window. The 64x512 f16 activation window is async-staged into
// LDS once (padded rows, bank-conflict free), then 8 waves cover the 96
// M-tiles of C[1536,64] = w_qkv_h * xn^T, 16 K-steps of WMMA each.
// mt/kt loops stay rolled so the LDS B-tiles are NOT hoisted+spilled; the
// epilogue is fully unrolled so acc[] uses static VGPR numbers (no movrels).
__global__ void __launch_bounds__(256)
qkv_kernel(const _Float16* __restrict__ wq, const _Float16* __restrict__ xn,
           _Float16* __restrict__ qh, _Float16* __restrict__ kh,
           _Float16* __restrict__ vt) {
  __shared__ __align__(16) _Float16 xs[PPW * XK_LD];  // 65 KB
  int n = blockIdx.x;
  const _Float16* bx = xn + (size_t)n * PPW * DIM;
  // 4096 b128 chunks: row = c>>6, 8-half chunk cc = c&63
  for (int c = threadIdx.x; c < PPW * (DIM / 8); c += 256) {
    int row = c >> 6, cc = c & 63;
    async_copy_b128(&xs[row * XK_LD + cc * 8], bx + (size_t)row * DIM + cc * 8);
  }
  wait_async();
  __syncthreads();

  int wave = threadIdx.x >> 5;
  int lane = threadIdx.x & 31;
#pragma unroll 1
  for (int mt = 0; mt < 12; ++mt) {
    int mtile = mt * 8 + wave;  // 0..95
    v8f acc[4] = {};
#pragma unroll 1
    for (int kt = 0; kt < 16; ++kt) {
      v16h a = load_a16x32(wq, DIM, lane, mtile * 16, kt * 32);
#pragma unroll
      for (int nt = 0; nt < 4; ++nt) {
        v16h b = load_b32x16(xs, XK_LD, lane, nt * 16, kt * 32);
        acc[nt] = wmma_f16(a, b, acc[nt]);
      }
    }
    // Epilogue, faithful to the torch .view of [nw, 512, 64] -> [nw,16,64,32].
    // Section (q/k/v) is uniform per M-tile: 32 tiles (512 rows) per section.
    int sec = mtile >> 5;              // 0=q 1=k 2=v (wave-uniform)
    int ebase = (mtile & 31) * 16;     // row within section
#pragma unroll
    for (int nt = 0; nt < 4; ++nt) {
#pragma unroll
      for (int r = 0; r < 8; ++r) {
        float val = acc[nt][r];
        int e = ebase + r + ((lane >> 4) << 3);  // 0..511 within section
        int pp = nt * 16 + (lane & 15);          // raw P index
        int h = e >> 5, rr = e & 31;
        int flat = rr * 64 + pp;
        int p = flat >> 5, c = flat & 31;
        size_t qk = (((size_t)n * HEADS + h) * PPW + p) * DH + c;
        if (sec == 0)      qh[qk] = (_Float16)(val * SCALE_Q);
        else if (sec == 1) kh[qk] = (_Float16)val;
        else vt[(((size_t)n * HEADS + h) * DH + c) * PPW + p] = (_Float16)val;
      }
    }
  }
}

// ---------------------------------------------------------- attention ------
// One block per (window, head); 4 waves x 16 rows. K and V^T are async-staged
// into padded LDS once per block. sim = QK^T (4 WMMA/wave, K=32 in one shot),
// softmax in C-layout with lane shuffles, attn staged to LDS f16 (padded),
// out = attn*V (4 WMMA/wave).
__global__ void __launch_bounds__(128)
attn_kernel(const _Float16* __restrict__ qh, const _Float16* __restrict__ kh,
            const _Float16* __restrict__ vt, const float* __restrict__ rel_bias,
            _Float16* __restrict__ ao) {
  __shared__ __align__(16) _Float16 ks[PPW * KK_LD];       // 64 rows x 32 (pad 40)
  __shared__ __align__(16) _Float16 vs[DH * VK_LD];        // 32 rows x 64 (pad 72)
  __shared__ __align__(16) _Float16 attn_lds[4][16][AK_LD];
  int nh = blockIdx.x;
  int n = nh >> 4, h = nh & 15;
  int wave = threadIdx.x >> 5;
  int lane = threadIdx.x & 31;
  const _Float16* qb = qh + ((size_t)n * HEADS + h) * PPW * DH;
  const _Float16* kb = kh + ((size_t)n * HEADS + h) * PPW * DH;
  const _Float16* vb = vt + ((size_t)n * HEADS + h) * DH * PPW;

  // stage K: 64 rows x 4 chunks; V: 32 rows x 8 chunks (8 halves per chunk)
  for (int c = threadIdx.x; c < PPW * (DH / 8); c += 128) {
    int row = c >> 2, cc = c & 3;
    async_copy_b128(&ks[row * KK_LD + cc * 8], kb + (size_t)row * DH + cc * 8);
  }
  for (int c = threadIdx.x; c < DH * (PPW / 8); c += 128) {
    int row = c >> 3, cc = c & 7;
    async_copy_b128(&vs[row * VK_LD + cc * 8], vb + (size_t)row * PPW + cc * 8);
  }
  v16h a = load_a16x32(qb, DH, lane, wave * 16, 0);  // Q overlaps the staging
  wait_async();
  __syncthreads();

  v8f sim[4] = {};
#pragma unroll
  for (int nt = 0; nt < 4; ++nt) {
    v16h b = load_b32x16(ks, KK_LD, lane, nt * 16, 0);
    sim[nt] = wmma_f16(a, b, sim[nt]);
  }
  // C layout: VGPR r, lane -> M = r + 8*(lane>>4), N = (lane&15). Row reduce
  // is a butterfly over the 16-lane half-groups (xor 1,2,4,8).
  int mbase = wave * 16 + ((lane >> 4) << 3);
  int col0 = lane & 15;
#pragma unroll
  for (int r = 0; r < 8; ++r) {
    float rowmax = -3.4e38f;
#pragma unroll
    for (int nt = 0; nt < 4; ++nt) {
      float v = sim[nt][r] + rel_bias[(mbase + r) * PPW + nt * 16 + col0];
      sim[nt][r] = v;
      rowmax = fmaxf(rowmax, v);
    }
    for (int msk = 1; msk <= 8; msk <<= 1)
      rowmax = fmaxf(rowmax, __shfl_xor(rowmax, msk, 32));
    float rowsum = 0.f;
#pragma unroll
    for (int nt = 0; nt < 4; ++nt) {
      float e = __expf(sim[nt][r] - rowmax);
      sim[nt][r] = e;
      rowsum += e;
    }
    for (int msk = 1; msk <= 8; msk <<= 1)
      rowsum += __shfl_xor(rowsum, msk, 32);
    float inv = 1.0f / rowsum;
#pragma unroll
    for (int nt = 0; nt < 4; ++nt)
      attn_lds[wave][r + ((lane >> 4) << 3)][nt * 16 + col0] =
          (_Float16)(sim[nt][r] * inv);
  }
  __syncthreads();
  v8f o[2] = {};
#pragma unroll
  for (int kt = 0; kt < 2; ++kt) {
    v16h aa = load_a16x32(&attn_lds[wave][0][0], AK_LD, lane, 0, kt * 32);
#pragma unroll
    for (int nt = 0; nt < 2; ++nt) {
      v16h b = load_b32x16(vs, VK_LD, lane, nt * 16, kt * 32);
      o[nt] = wmma_f16(aa, b, o[nt]);
    }
  }
  // out layout for projection B: ao[n][p][i], i = h*32 + c (i contiguous)
#pragma unroll
  for (int nt = 0; nt < 2; ++nt)
#pragma unroll
    for (int r = 0; r < 8; ++r) {
      int p = wave * 16 + r + ((lane >> 4) << 3);
      int c = nt * 16 + (lane & 15);
      ao[((size_t)n * PPW + p) * DIM + h * DH + c] = (_Float16)o[nt][r];
    }
}

// ------------------------------------------------------ output projection --
// One block per window; attention output window async-staged into LDS once,
// 8 waves cover the 32 M-tiles of C[512,64] = w_out_h * ao^T.
// mt/kt loops stay rolled; epilogue fully unrolled (static acc indexing).
__global__ void __launch_bounds__(256)
proj_kernel(const _Float16* __restrict__ wo, const _Float16* __restrict__ ao,
            const float* __restrict__ bout, float* __restrict__ out) {
  __shared__ __align__(16) _Float16 as[PPW * XK_LD];  // 65 KB
  int n = blockIdx.x;
  const _Float16* bx = ao + (size_t)n * PPW * DIM;
  for (int c = threadIdx.x; c < PPW * (DIM / 8); c += 256) {
    int row = c >> 6, cc = c & 63;
    async_copy_b128(&as[row * XK_LD + cc * 8], bx + (size_t)row * DIM + cc * 8);
  }
  wait_async();
  __syncthreads();

  int wave = threadIdx.x >> 5;
  int lane = threadIdx.x & 31;
  int bb = n >> 6, wy = (n >> 3) & 7, wx = n & 7;
#pragma unroll 1
  for (int mt = 0; mt < 4; ++mt) {
    int mtile = mt * 8 + wave;  // 0..31
    v8f acc[4] = {};
#pragma unroll 1
    for (int kt = 0; kt < 16; ++kt) {
      v16h a = load_a16x32(wo, DIM, lane, mtile * 16, kt * 32);
#pragma unroll
      for (int nt = 0; nt < 4; ++nt) {
        v16h b = load_b32x16(as, XK_LD, lane, nt * 16, kt * 32);
        acc[nt] = wmma_f16(a, b, acc[nt]);
      }
    }
#pragma unroll
    for (int nt = 0; nt < 4; ++nt)
#pragma unroll
      for (int r = 0; r < 8; ++r) {
        int o = mtile * 16 + r + ((lane >> 4) << 3);
        int p = nt * 16 + (lane & 15);
        int py = p >> 3, px = p & 7;
        out[(((size_t)bb * DIM + o) * IMG + wy * 8 + py) * IMG + wx * 8 + px] =
            acc[nt][r] + bout[o];
      }
  }
}

extern "C" void kernel_launch(void* const* d_in, const int* in_sizes, int n_in,
                              void* d_out, int out_size, void* d_ws, size_t ws_size,
                              hipStream_t stream) {
  const float* x      = (const float*)d_in[0];
  const float* g      = (const float*)d_in[1];
  const float* bch    = (const float*)d_in[2];
  const float* w_qkv  = (const float*)d_in[3];
  const float* w_out  = (const float*)d_in[4];
  const float* b_out  = (const float*)d_in[5];
  const float* dpb_w1 = (const float*)d_in[6];
  const float* dpb_b1 = (const float*)d_in[7];
  const float* dpb_g1 = (const float*)d_in[8];
  const float* dpb_be1= (const float*)d_in[9];
  const float* dpb_w2 = (const float*)d_in[10];
  const float* dpb_b2 = (const float*)d_in[11];
  const float* dpb_g2 = (const float*)d_in[12];
  const float* dpb_be2= (const float*)d_in[13];
  const float* dpb_w3 = (const float*)d_in[14];
  const float* dpb_b3 = (const float*)d_in[15];
  float* out = (float*)d_out;

  char* ws = (char*)d_ws;
  size_t off = 0;
  auto carve = [&](size_t bytes) -> void* {
    void* p = ws + off;
    off = (off + bytes + 255) & ~(size_t)255;
    return p;
  };
  float*    rel_bias = (float*)   carve((size_t)PPW * PPW * 4);
  _Float16* wq_h     = (_Float16*)carve((size_t)3 * DIM * DIM * 2);
  _Float16* wo_h     = (_Float16*)carve((size_t)DIM * DIM * 2);
  _Float16* xn       = (_Float16*)carve((size_t)NWIN * PPW * DIM * 2);
  _Float16* qh       = (_Float16*)carve((size_t)NWIN * HEADS * PPW * DH * 2);
  _Float16* kh       = (_Float16*)carve((size_t)NWIN * HEADS * PPW * DH * 2);
  _Float16* vt       = (_Float16*)carve((size_t)NWIN * HEADS * DH * PPW * 2);
  _Float16* ao       = (_Float16*)carve((size_t)NWIN * PPW * DIM * 2);

  dpb_kernel<<<1, 512, 0, stream>>>(dpb_w1, dpb_b1, dpb_g1, dpb_be1,
                                    dpb_w2, dpb_b2, dpb_g2, dpb_be2,
                                    dpb_w3, dpb_b3, rel_bias);
  cvt_f16<<<(3 * DIM * DIM) / 256, 256, 0, stream>>>(w_qkv, wq_h, 3 * DIM * DIM);
  cvt_f16<<<(DIM * DIM) / 256, 256, 0, stream>>>(w_out, wo_h, DIM * DIM);
  ln_kernel<<<32768 / 256, 256, 0, stream>>>(x, g, bch, xn);
  qkv_kernel<<<NWIN, 256, 0, stream>>>(wq_h, xn, qh, kh, vt);
  attn_kernel<<<NWIN * HEADS, 128, 0, stream>>>(qh, kh, vt, rel_bias, ao);
  proj_kernel<<<NWIN, 256, 0, stream>>>(wo_h, ao, b_out, out);
}
